// GATRecommender_85813446574384
// MI455X (gfx1250) — compile-verified, matrix operand order checked by reading, not run on hardware
//
#include <hip/hip_runtime.h>
#include <hip/hip_bf16.h>
#include <stdint.h>

// ---------------------------------------------------------------------------
// Types for CDNA5 WMMA (wave32): v16bf A/B fragments, v8f f32 accumulator
// ---------------------------------------------------------------------------
typedef __bf16 v16bf __attribute__((ext_vector_type(16)));
typedef float  v8f   __attribute__((ext_vector_type(8)));
typedef int    v4i   __attribute__((ext_vector_type(4)));

typedef __attribute__((address_space(1))) v4i* gptr_v4i;   // global
typedef __attribute__((address_space(3))) v4i* lptr_v4i;   // LDS

union Frag {
  v16bf v;
  uint4 q[2];
};

// ---------------------------------------------------------------------------
// CDNA5 async copy (global -> LDS, 16 bytes), tracked by ASYNCcnt
// ---------------------------------------------------------------------------
__device__ __forceinline__ void async_copy16(const void* g, void* l) {
#if __has_builtin(__builtin_amdgcn_global_load_async_to_lds_b128)
  __builtin_amdgcn_global_load_async_to_lds_b128(
      (gptr_v4i)(uintptr_t)g, (lptr_v4i)(uintptr_t)l, 0, 0);
#else
  unsigned lds_off = (unsigned)(uintptr_t)l;   // generic shared addr: low 32 = LDS offset
  asm volatile("global_load_async_to_lds_b128 %0, %1, off"
               :: "v"(lds_off), "v"(g) : "memory");
#endif
}

__device__ __forceinline__ void wait_async0() {
#if __has_builtin(__builtin_amdgcn_s_wait_asynccnt)
  __builtin_amdgcn_s_wait_asynccnt(0);
#else
  asm volatile("s_wait_asynccnt 0x0" ::: "memory");
#endif
}

// ---------------------------------------------------------------------------
// Helpers
// ---------------------------------------------------------------------------
__device__ __forceinline__ unsigned short f32_to_bf16(float f) {
  union { float f; uint32_t u; } c; c.f = f;
  uint32_t u = c.u;
  uint32_t r = u + 0x7FFFu + ((u >> 16) & 1u);   // round-to-nearest-even
  return (unsigned short)(r >> 16);
}

// Order-preserving float -> uint key (for atomicMax-based segment max)
__device__ __forceinline__ unsigned fkey(float f) {
  union { float f; unsigned u; } c; c.f = f;
  return (c.u & 0x80000000u) ? ~c.u : (c.u | 0x80000000u);
}
__device__ __forceinline__ float funkey(unsigned k) {
  union { float f; unsigned u; } c;
  c.u = (k & 0x80000000u) ? (k & 0x7FFFFFFFu) : ~k;
  return c.f;
}

__device__ __forceinline__ void edge_sd(const int* __restrict__ ei, long long E,
                                        long long i, int& s, int& d) {
  if (i < E) { s = ei[i]; d = ei[E + i]; }
  else       { s = d = (int)(i - E); }      // self loop
}

// ---------------------------------------------------------------------------
// K0: elementwise f32 -> bf16 convert
// ---------------------------------------------------------------------------
__global__ __launch_bounds__(256) void cvt_bf16_kernel(
    const float* __restrict__ in, unsigned short* __restrict__ out, long long n) {
  long long gid = (long long)blockIdx.x * blockDim.x + threadIdx.x;
  if (gid < n) out[gid] = f32_to_bf16(in[gid]);
}

// ---------------------------------------------------------------------------
// K1: tiled transpose + convert: in [R,C] f32 row-major -> out [C,R] bf16.
// R, C multiples of 32.
// ---------------------------------------------------------------------------
__global__ __launch_bounds__(256) void transpose_cvt_bf16(
    const float* __restrict__ in, unsigned short* __restrict__ out, int R, int C) {
  __shared__ unsigned short t[32][33];
  int tx = threadIdx.x & 31;
  int ty = threadIdx.x >> 5;              // 0..7
  int ctiles = C >> 5;
  int ct = blockIdx.x % ctiles;
  int rt = blockIdx.x / ctiles;
  int c0 = ct << 5, r0 = rt << 5;
#pragma unroll
  for (int i = 0; i < 32; i += 8)
    t[ty + i][tx] = f32_to_bf16(in[(size_t)(r0 + ty + i) * C + c0 + tx]);
  __syncthreads();
#pragma unroll
  for (int i = 0; i < 32; i += 8)
    out[(size_t)(c0 + ty + i) * R + r0 + tx] = t[tx][ty + i];
}

// ---------------------------------------------------------------------------
// K2: u32 pattern fill (for -inf max-keys)
// ---------------------------------------------------------------------------
__global__ __launch_bounds__(256) void fill_u32_kernel(
    unsigned* __restrict__ p, unsigned val, long long n) {
  long long gid = (long long)blockIdx.x * blockDim.x + threadIdx.x;
  if (gid < n) p[gid] = val;
}

// ---------------------------------------------------------------------------
// K3: bf16 WMMA GEMM with async-LDS-staged B.
//   C[M,Ncols] = A[M,K] * B (+ bias), f32 accumulate.
//   A  : [M, K]      bf16 row-major
//   Bt : [Ncols, K]  bf16 (B transposed; each output column contiguous)
// Workgroup = 8 waves: one 16-col n-tile, 8 consecutive 16-row m-tiles.
// The shared 16xK B-slab is staged once into LDS via async copies
// (ASYNCcnt), rows padded by 8 halfs for conflict-free ds_load_b128.
// M, Ncols mult of 16; K mult of 32; K <= 256.
// A fragment per lane L: row m0+(L&15); 16 K-values at k0+(L>>4)*8 (+0..7)
// and k0+16+(L>>4)*8 (+0..7) -> two 16B loads. B fragment symmetric.
// ---------------------------------------------------------------------------
#define BPAD 8
__global__ __launch_bounds__(256) void wmma_gemm_bf16_lds(
    const unsigned short* __restrict__ A, const unsigned short* __restrict__ Bt,
    float* __restrict__ C, const float* __restrict__ bias,
    int M, int Ncols, int K) {
  __shared__ __align__(16) unsigned short Bs[16 * (256 + BPAD)];

  int lane = threadIdx.x & 31;
  int wv   = threadIdx.x >> 5;               // wave in block: 0..7
  int ntN  = Ncols >> 4;
  int Mtiles = M >> 4;
  int bm = blockIdx.x / ntN;                 // m-block (8 m-tiles)
  int nt = blockIdx.x % ntN;
  int n0 = nt << 4;
  int rowStride = K + BPAD;

  // ---- stage B slab: Bt[n0 .. n0+15][0..K) -> LDS (async, ASYNCcnt) ----
  {
    const unsigned short* bBase = Bt + (size_t)n0 * K;
    int chunksPerRow = K >> 3;               // 16B chunks per row
    int totalChunks  = chunksPerRow << 4;    // 16 rows
    for (int t = threadIdx.x; t < totalChunks; t += 256) {
      int r = t / chunksPerRow, cc = t - r * chunksPerRow;
      async_copy16(bBase + (size_t)r * K + cc * 8,
                   &Bs[r * rowStride + cc * 8]);
    }
  }
  wait_async0();
  __syncthreads();

  int mt = bm * 8 + wv;
  if (mt >= Mtiles) return;                  // whole-wave uniform; no more barriers
  int m0 = mt << 4;
  int half = lane >> 4;                      // 0 or 1
  int lrow = lane & 15;

  const unsigned short* aRow = A + (size_t)(m0 + lrow) * K + half * 8;
  const unsigned short* bRow = &Bs[lrow * rowStride + half * 8];

  v8f acc = {0.f, 0.f, 0.f, 0.f, 0.f, 0.f, 0.f, 0.f};
  for (int k0 = 0; k0 < K; k0 += 32) {
    Frag a, b;
    a.q[0] = *(const uint4*)(aRow + k0);
    a.q[1] = *(const uint4*)(aRow + k0 + 16);
    b.q[0] = *(const uint4*)(bRow + k0);     // ds_load_b128, conflict-free
    b.q[1] = *(const uint4*)(bRow + k0 + 16);
    if (k0 + 32 < K)                         // -> global_prefetch_b8
      __builtin_prefetch(aRow + k0 + 32, 0, 1);
    acc = __builtin_amdgcn_wmma_f32_16x16x32_bf16(
        false, a.v, false, b.v, (short)0, acc, false, false);
  }

  int col = n0 + lrow;
  float bv = bias ? bias[col] : 0.0f;
  float* cP = C + (size_t)(m0 + half * 8) * Ncols + col;
#pragma unroll
  for (int v = 0; v < 8; ++v)                // row m0 + half*8 + v
    cP[(size_t)v * Ncols] = acc[v] + bv;
}

// ---------------------------------------------------------------------------
// K4: attention scores  a_s[n,h] = <h[n,h,:], att_s[h,:]>, same for a_d
// ---------------------------------------------------------------------------
__global__ __launch_bounds__(256) void gat_scores_kernel(
    const float* __restrict__ h, const float* __restrict__ att_s,
    const float* __restrict__ att_d, float* __restrict__ as_o,
    float* __restrict__ ad_o, int N, int H, int Cc) {
  long long gid = (long long)blockIdx.x * blockDim.x + threadIdx.x;
  if (gid >= (long long)N * H) return;
  int n = (int)(gid / H), hh = (int)(gid % H);
  const float* row = h + (size_t)n * H * Cc + (size_t)hh * Cc;
  const float* as = att_s + hh * Cc;
  const float* ad = att_d + hh * Cc;
  float s = 0.f, d = 0.f;
  for (int c = 0; c < Cc; c += 4) {
    float4 v = *(const float4*)(row + c);
    float4 a = *(const float4*)(as + c);
    float4 b = *(const float4*)(ad + c);
    s += v.x * a.x + v.y * a.y + v.z * a.z + v.w * a.w;
    d += v.x * b.x + v.y * b.y + v.z * b.z + v.w * b.w;
  }
  as_o[gid] = s; ad_o[gid] = d;
}

// ---------------------------------------------------------------------------
// K5: pass 1 — e = leaky_relu(a_s[src]+a_d[dst]); segment max via keyed atomicMax
// ---------------------------------------------------------------------------
__global__ __launch_bounds__(256) void edge_max_kernel(
    const float* __restrict__ as_, const float* __restrict__ ad_,
    const int* __restrict__ ei, long long E, long long Etot, int H,
    float* __restrict__ ebuf, unsigned* __restrict__ maxk) {
  long long gid = (long long)blockIdx.x * blockDim.x + threadIdx.x;
  if (gid >= Etot * H) return;
  long long e = gid / H; int hh = (int)(gid % H);
  int s, d; edge_sd(ei, E, e, s, d);
  float v = as_[(size_t)s * H + hh] + ad_[(size_t)d * H + hh];
  v = v > 0.f ? v : 0.2f * v;               // negative_slope = 0.2
  ebuf[gid] = v;
  atomicMax(maxk + (size_t)d * H + hh, fkey(v));
}

// ---------------------------------------------------------------------------
// K6: pass 2 — ex = exp(e - max[dst]); segment sum via f32 atomics
// ---------------------------------------------------------------------------
__global__ __launch_bounds__(256) void edge_expsum_kernel(
    const int* __restrict__ ei, long long E, long long Etot, int H,
    const unsigned* __restrict__ maxk, float* __restrict__ ebuf,
    float* __restrict__ sum) {
  long long gid = (long long)blockIdx.x * blockDim.x + threadIdx.x;
  if (gid >= Etot * H) return;
  long long e = gid / H; int hh = (int)(gid % H);
  int s, d; edge_sd(ei, E, e, s, d);
  float m = funkey(maxk[(size_t)d * H + hh]);
  float ex = expf(ebuf[gid] - m);
  ebuf[gid] = ex;
  unsafeAtomicAdd(sum + (size_t)d * H + hh, ex);
}

// ---------------------------------------------------------------------------
// K7: pass 3 — out[dst] += alpha * h[src]; (H*Cc)/4 threads per edge, float4
// ---------------------------------------------------------------------------
__global__ __launch_bounds__(256) void edge_aggr_kernel(
    const float* __restrict__ ebuf, const float* __restrict__ sum,
    const float* __restrict__ hmat, const int* __restrict__ ei,
    long long E, long long Etot, int H, int Cc, float* __restrict__ out) {
  int HC = H * Cc;
  int tpe = HC >> 2;
  long long gid = (long long)blockIdx.x * blockDim.x + threadIdx.x;
  if (gid >= Etot * tpe) return;
  long long e = gid / tpe; int slot = (int)(gid % tpe);
  int c4 = slot << 2;
  int hh = c4 / Cc;
  int s, d; edge_sd(ei, E, e, s, d);
  float alpha = ebuf[(size_t)e * H + hh] /
                (sum[(size_t)d * H + hh] + 1e-16f);
  float4 hv = *(const float4*)(hmat + (size_t)s * HC + c4);
  float* o = out + (size_t)d * HC + c4;
  unsafeAtomicAdd(o + 0, hv.x * alpha);
  unsafeAtomicAdd(o + 1, hv.y * alpha);
  unsafeAtomicAdd(o + 2, hv.z * alpha);
  unsafeAtomicAdd(o + 3, hv.w * alpha);
}

// ---------------------------------------------------------------------------
// K8: bias (+ optional ELU) then convert to bf16
// ---------------------------------------------------------------------------
__global__ __launch_bounds__(256) void bias_act_cvt_kernel(
    const float* __restrict__ in, const float* __restrict__ bias,
    unsigned short* __restrict__ out, long long n, int cmask, int do_elu) {
  long long gid = (long long)blockIdx.x * blockDim.x + threadIdx.x;
  if (gid >= n) return;
  float v = in[gid] + bias[(int)gid & cmask];
  if (do_elu) v = v > 0.f ? v : (expf(v) - 1.0f);
  out[gid] = f32_to_bf16(v);
}

// ---------------------------------------------------------------------------
// K9: build x[b, 0:64] = user_emb[user_idx[b]]; x[b, 64+j*64+c] = g2[ctx[b,j], c]
// Output bf16 [B, D*(CTX+1)].
// ---------------------------------------------------------------------------
__global__ __launch_bounds__(256) void build_x_kernel(
    const float* __restrict__ user_emb, const int* __restrict__ user_idx,
    const int* __restrict__ ctx_idx, const unsigned short* __restrict__ g2bf,
    unsigned short* __restrict__ xbf, int B, int CTX, int D) {
  int KF = D * (CTX + 1);
  long long gid = (long long)blockIdx.x * blockDim.x + threadIdx.x;
  if (gid >= (long long)B * KF) return;
  int b = (int)(gid / KF), col = (int)(gid % KF);
  if (col < D) {
    xbf[gid] = f32_to_bf16(user_emb[(size_t)user_idx[b] * D + col]);
  } else {
    int j = (col - D) / D, c = col % D;
    xbf[gid] = g2bf[(size_t)ctx_idx[b * CTX + j] * D + c];
  }
}

// ---------------------------------------------------------------------------
// Host-side launch
// ---------------------------------------------------------------------------
static inline int cdiv_ll(long long a, long long b) { return (int)((a + b - 1) / b); }

extern "C" void kernel_launch(void* const* d_in, const int* in_sizes, int n_in,
                              void* d_out, int out_size, void* d_ws, size_t ws_size,
                              hipStream_t stream) {
  const int*   user_idx    = (const int*)  d_in[0];
  const int*   context_idx = (const int*)  d_in[1];
  const int*   edge_index  = (const int*)  d_in[2];
  const float* user_emb    = (const float*)d_in[3];
  const float* service_emb = (const float*)d_in[4];
  const float* W1          = (const float*)d_in[5];
  const float* att_src1    = (const float*)d_in[6];
  const float* att_dst1    = (const float*)d_in[7];
  const float* b1          = (const float*)d_in[8];
  const float* W2          = (const float*)d_in[9];
  const float* att_src2    = (const float*)d_in[10];
  const float* att_dst2    = (const float*)d_in[11];
  const float* b2          = (const float*)d_in[12];
  const float* fc_w        = (const float*)d_in[13];
  const float* fc_b        = (const float*)d_in[14];

  const int D = 64, H = 2, CTX = 3;
  const int B  = in_sizes[0];
  const long long E = in_sizes[2] / 2;
  const int N  = in_sizes[4] / D;           // service nodes
  const int NS = in_sizes[14];              // num services (fc_b)
  const long long Etot = E + N;
  const int K2  = H * D;                    // 128
  const int KFC = D * (CTX + 1);            // 256
  const int TB = 256;

  // --- workspace carve-out (256B aligned) ---
  char* base = (char*)d_ws; size_t off = 0;
  auto walloc = [&](size_t bytes) -> void* {
    void* p = base + off;
    off = (off + bytes + 255) & ~(size_t)255;
    return p;
  };
  unsigned short* se_bf = (unsigned short*)walloc((size_t)N * D * 2);   // also g2bf later
  unsigned short* w1t   = (unsigned short*)walloc((size_t)K2 * D * 2);  // [128,64]
  float*          h1    = (float*)walloc((size_t)N * K2 * 4);           // also h2 later
  float*          as1   = (float*)walloc((size_t)N * H * 4);
  float*          ad1   = (float*)walloc((size_t)N * H * 4);
  float*          ebuf  = (float*)walloc((size_t)Etot * H * 4);
  unsigned*       maxk  = (unsigned*)walloc((size_t)N * H * 4);
  float*          sum   = (float*)walloc((size_t)N * H * 4);
  float*          out1  = (float*)walloc((size_t)N * K2 * 4);           // also out2 later
  unsigned short* g1bf  = (unsigned short*)walloc((size_t)N * K2 * 2);
  unsigned short* w2t   = (unsigned short*)walloc((size_t)D * K2 * 2);  // [64,128]
  unsigned short* xbf   = (unsigned short*)walloc((size_t)B * KFC * 2);
  unsigned short* fcwt  = (unsigned short*)walloc((size_t)NS * KFC * 2);// [NS,256]
  float*          h2    = h1;          // alias: h1 dead after layer-1 aggregation
  float*          out2  = out1;        // re-zeroed before use
  float*          as2   = as1;
  float*          ad2   = ad1;
  unsigned short* g2bf  = se_bf;       // alias: se_bf dead after GEMM1
  (void)ws_size; (void)n_in; (void)out_size;

  const unsigned NEG_INF_KEY = 0x007FFFFFu; // fkey(-inf)

  auto gemm_grid = [](int M, int Ncols) -> int {
    int mtiles = M >> 4;
    int mblocks = (mtiles + 7) >> 3;        // 8 m-tiles (waves) per block
    return mblocks * (Ncols >> 4);
  };

  // ===== Layer 1 =====
  cvt_bf16_kernel<<<cdiv_ll((long long)N * D, TB), TB, 0, stream>>>(
      service_emb, se_bf, (long long)N * D);
  transpose_cvt_bf16<<<(D / 32) * (K2 / 32), TB, 0, stream>>>(W1, w1t, D, K2);
  wmma_gemm_bf16_lds<<<gemm_grid(N, K2), TB, 0, stream>>>(
      se_bf, w1t, h1, nullptr, N, K2, D);
  gat_scores_kernel<<<cdiv_ll((long long)N * H, TB), TB, 0, stream>>>(
      h1, att_src1, att_dst1, as1, ad1, N, H, D);

  fill_u32_kernel<<<cdiv_ll((long long)N * H, TB), TB, 0, stream>>>(
      maxk, NEG_INF_KEY, (long long)N * H);
  (void)hipMemsetAsync(sum, 0, (size_t)N * H * 4, stream);
  (void)hipMemsetAsync(out1, 0, (size_t)N * K2 * 4, stream);

  edge_max_kernel<<<cdiv_ll(Etot * H, TB), TB, 0, stream>>>(
      as1, ad1, edge_index, E, Etot, H, ebuf, maxk);
  edge_expsum_kernel<<<cdiv_ll(Etot * H, TB), TB, 0, stream>>>(
      edge_index, E, Etot, H, maxk, ebuf, sum);
  edge_aggr_kernel<<<cdiv_ll(Etot * (H * D / 4), TB), TB, 0, stream>>>(
      ebuf, sum, h1, edge_index, E, Etot, H, D, out1);
  bias_act_cvt_kernel<<<cdiv_ll((long long)N * K2, TB), TB, 0, stream>>>(
      out1, b1, g1bf, (long long)N * K2, K2 - 1, /*elu=*/1);

  // ===== Layer 2 (H=1) =====
  transpose_cvt_bf16<<<(K2 / 32) * (D / 32), TB, 0, stream>>>(W2, w2t, K2, D);
  wmma_gemm_bf16_lds<<<gemm_grid(N, D), TB, 0, stream>>>(
      g1bf, w2t, h2, nullptr, N, D, K2);
  gat_scores_kernel<<<cdiv_ll((long long)N, TB), TB, 0, stream>>>(
      h2, att_src2, att_dst2, as2, ad2, N, 1, D);

  fill_u32_kernel<<<cdiv_ll((long long)N, TB), TB, 0, stream>>>(
      maxk, NEG_INF_KEY, (long long)N);
  (void)hipMemsetAsync(sum, 0, (size_t)N * 4, stream);
  (void)hipMemsetAsync(out2, 0, (size_t)N * D * 4, stream);

  edge_max_kernel<<<cdiv_ll(Etot, TB), TB, 0, stream>>>(
      as2, ad2, edge_index, E, Etot, 1, ebuf, maxk);
  edge_expsum_kernel<<<cdiv_ll(Etot, TB), TB, 0, stream>>>(
      edge_index, E, Etot, 1, maxk, ebuf, sum);
  edge_aggr_kernel<<<cdiv_ll(Etot * (D / 4), TB), TB, 0, stream>>>(
      ebuf, sum, h2, edge_index, E, Etot, 1, D, out2);
  bias_act_cvt_kernel<<<cdiv_ll((long long)N * D, TB), TB, 0, stream>>>(
      out2, b2, g2bf, (long long)N * D, D - 1, /*elu=*/0);

  // ===== Final FC: [B, KFC] @ [KFC, NS] + fc_b -> d_out =====
  build_x_kernel<<<cdiv_ll((long long)B * KFC, TB), TB, 0, stream>>>(
      user_emb, user_idx, context_idx, g2bf, xbf, B, CTX, D);
  transpose_cvt_bf16<<<(KFC / 32) * (NS / 32), TB, 0, stream>>>(fc_w, fcwt, KFC, NS);
  wmma_gemm_bf16_lds<<<gemm_grid(B, NS), TB, 0, stream>>>(
      xbf, fcwt, (float*)d_out, fc_b, B, NS, KFC);
}
